// ResRNN_6932077216203
// MI455X (gfx1250) — matrix operand order repaired and checked by reading, not compiled, not run on hardware
//
#include <hip/hip_runtime.h>
#include <hip/hip_bf16.h>

// ---------------------------------------------------------------------------
// ResRNN for MI455X (gfx1250): bf16 weights resident in 192MB L2,
// WMMA (v_wmma_f32_16x16x32_bf16) for both batched GEMMs and the
// sequential per-step GEMVs, persistent kernel with device-wide barriers.
// ---------------------------------------------------------------------------

typedef __bf16 bf16_t;
typedef __attribute__((ext_vector_type(16))) __bf16 v16bf;
typedef __attribute__((ext_vector_type(8)))  __bf16 v8bf;
typedef __attribute__((ext_vector_type(8)))  float   v8f;
typedef __attribute__((ext_vector_type(4)))  float   v4f;

constexpr int IN_DIMC  = 1024;
constexpr int H_DIMC   = 2048;
constexpr int H2C      = 4096;   // 2*H
constexpr int OUT_DIMC = 1024;
constexpr int T_STEPS  = 512;
constexpr int SCAN_WGS = 128;    // persistent workgroups (256 thr = 8 waves each)

// ---- workspace layout (bytes) ----
constexpr size_t OFF_INPW = 0;                                         // 2048x1024 bf16
constexpr size_t OFF_L1W  = OFF_INPW + (size_t)H_DIMC*IN_DIMC*2;       // 4096x4096 bf16
constexpr size_t OFF_L2W  = OFF_L1W  + (size_t)H2C*H2C*2;              // 4096x4096 bf16
constexpr size_t OFF_XHW  = OFF_L2W  + (size_t)H2C*H2C*2;              // 2048x4096 bf16
constexpr size_t OFF_OUTW = OFF_XHW  + (size_t)H_DIMC*H2C*2;           // 1024x2048 bf16
constexpr size_t OFF_X    = OFF_OUTW + (size_t)OUT_DIMC*H_DIMC*2;      // 512x1024 bf16
constexpr size_t OFF_XI   = OFF_X    + (size_t)T_STEPS*IN_DIMC*2;      // 512x2048 bf16
constexpr size_t OFF_HALL = OFF_XI   + (size_t)T_STEPS*H_DIMC*2;       // 512x2048 bf16
constexpr size_t OFF_U1   = OFF_HALL + (size_t)T_STEPS*H_DIMC*2;       // 4096 bf16
constexpr size_t OFF_TMP  = OFF_U1   + (size_t)H2C*2;                  // 4096 bf16
constexpr size_t OFF_HB   = OFF_TMP  + (size_t)H2C*2;                  // 2048 bf16
constexpr size_t OFF_SYNC = OFF_HB   + (size_t)H_DIMC*2;               // 2 uints

// ---------------------------------------------------------------------------
// helpers
// ---------------------------------------------------------------------------
__device__ inline v8f bf16_mma(v16bf a, v16bf b, v8f c) {
    return __builtin_amdgcn_wmma_f32_16x16x32_bf16(
        /*neg_a=*/false, a, /*neg_b=*/false, b,
        /*c_mod=*/(short)0, c, /*reuse_a=*/false, /*reuse_b=*/false);
}

// A (16x32 bf16, row-major source): lane<16 -> row=lane, K {kb..kb+15}
//                                   lane>=16 -> row=lane-16, K {kb+16..kb+31}
__device__ inline v16bf load_a_tile(const bf16_t* arow, int kb, int koffA) {
    v8bf lo = *(const v8bf*)(arow + kb + 2 * koffA);
    v8bf hi = *(const v8bf*)(arow + kb + 2 * koffA + 8);
    return __builtin_shufflevector(lo, hi, 0,1,2,3,4,5,6,7,8,9,10,11,12,13,14,15);
}

// device-wide barrier (persistent kernel; SCAN_WGS blocks all resident)
__device__ inline void grid_barrier(unsigned* cnt, unsigned* gen, unsigned nwg) {
    __syncthreads();
    if (threadIdx.x == 0) {
        __threadfence();
        volatile unsigned* vgen = gen;
        unsigned g = *vgen;
        if (atomicAdd(cnt, 1u) == nwg - 1u) {
            *cnt = 0u;
            __threadfence();
            *vgen = g + 1u;
        } else {
            while (*vgen == g) { __builtin_amdgcn_s_sleep(1); }
        }
        __threadfence();
    }
    __syncthreads();
}

// ---------------------------------------------------------------------------
// phase 0: fp32 -> bf16 conversion, and init
// ---------------------------------------------------------------------------
__global__ __launch_bounds__(256) void cvt_kernel(const float* __restrict__ src,
                                                  bf16_t* __restrict__ dst, int n) {
    int i = blockIdx.x * blockDim.x + threadIdx.x;
    int stride = gridDim.x * blockDim.x;
    for (; i < n; i += stride) dst[i] = (bf16_t)src[i];
}

__global__ __launch_bounds__(256) void init_kernel(const float* __restrict__ h,
                                                   bf16_t* __restrict__ hbuf,
                                                   unsigned* __restrict__ sync) {
    int i = blockIdx.x * blockDim.x + threadIdx.x;
    if (i < H_DIMC) hbuf[i] = (bf16_t)h[i];
    if (i == 0) { sync[0] = 0u; sync[1] = 0u; }
}

// ---------------------------------------------------------------------------
// phase 1: XI[t, j] = inp_w[j,:] . x[t,:] + inp_b[j]   (batched GEMM, WMMA)
// A = inp_w bf16 [2048 x 1024], B[k][t] = xbf[t*1024 + k], out bf16 [T x 2048]
// ---------------------------------------------------------------------------
__global__ __launch_bounds__(256) void xi_gemm(const bf16_t* __restrict__ wbf,
                                               const bf16_t* __restrict__ xbf,
                                               const float*  __restrict__ bias,
                                               bf16_t* __restrict__ xi) {
    const int lane  = threadIdx.x & 31;
    const int gw    = (blockIdx.x * blockDim.x + threadIdx.x) >> 5;
    const int nwv   = (gridDim.x * blockDim.x) >> 5;
    const int mrow  = lane & 15;
    const int koffA = (lane & 16) ? 8 : 0;
    const int koffB = (lane & 16) ? 16 : 0;
    constexpr int MT = H_DIMC / 16, NT = T_STEPS / 16;

    for (int tile = gw; tile < MT * NT; tile += nwv) {
        const int mt = tile / NT, nt = tile % NT;
        const bf16_t* arow = wbf + (size_t)(mt * 16 + mrow) * IN_DIMC;
        const bf16_t* bcol = xbf + (size_t)(nt * 16 + mrow) * IN_DIMC;  // col = lane&15
        v8f c = {};
        for (int kb = 0; kb < IN_DIMC; kb += 32) {
            v16bf a = load_a_tile(arow, kb, koffA);
            v16bf b = *(const v16bf*)(bcol + kb + koffB);
            c = bf16_mma(a, b, c);
        }
        const int tcol = nt * 16 + (lane & 15);
        const int j0   = mt * 16 + ((lane & 16) ? 8 : 0);
        v8bf o;
        for (int r = 0; r < 8; ++r) o[r] = (bf16_t)(c[r] + bias[j0 + r]);
        *(v8bf*)(xi + (size_t)tcol * H_DIMC + j0) = o;
    }
}

// ---------------------------------------------------------------------------
// phase 2: persistent sequential scan. Per step:
//   A) u1   = relu(l1_w @ [xi_t, h] + l1_b)
//   B) tmp  = relu([xi_t, h] + l2_w @ u1 + l2_b)
//   C) h    = xh2h_w @ tmp + xh2h_b   (store bf16 into Hall[t], update hbuf)
// Each WG owns row tiles; its 8 waves split K=4096, LDS-reduce (deterministic).
// 3 grid barriers / step.
// ---------------------------------------------------------------------------
__global__ __launch_bounds__(256, 1) void scan_kernel(
        const bf16_t* __restrict__ l1w, const bf16_t* __restrict__ l2w,
        const bf16_t* __restrict__ xhw,
        const float* __restrict__ l1b, const float* __restrict__ l2b,
        const float* __restrict__ xhb,
        const bf16_t* __restrict__ xi,     // [T x 2048]
        bf16_t* __restrict__ hall,         // [T x 2048]
        bf16_t* __restrict__ u1, bf16_t* __restrict__ tmpv,
        bf16_t* __restrict__ hbuf,         // [2048]
        float* __restrict__ hfinal,        // d_out + T*OUT
        unsigned* __restrict__ sync) {
    __shared__ float red[8][16];
    const int tid   = threadIdx.x;
    const int wave  = tid >> 5;
    const int lane  = tid & 31;
    const int mrow  = lane & 15;
    const int koffA = (lane & 16) ? 8 : 0;
    const int koffB = (lane & 16) ? 16 : 0;
    const int kb0   = wave * (H2C / 8);    // per-wave K range, 512 wide
    const unsigned nwg = gridDim.x;
    unsigned* cnt = sync;
    unsigned* gen = sync + 1;

    for (int t = 0; t < T_STEPS; ++t) {
        const bf16_t* xivec = xi + (size_t)t * H_DIMC;

        // ---- A: l1 ----
        for (int tile = blockIdx.x; tile < H2C / 16; tile += nwg) {
            const int mbase = tile * 16;
            const bf16_t* arow = l1w + (size_t)(mbase + mrow) * H2C;
            v8f c = {};
            for (int kb = kb0; kb < kb0 + H2C / 8; kb += 32) {
                v16bf a = load_a_tile(arow, kb, koffA);
                const bf16_t* bsrc = (kb < H_DIMC) ? (xivec + kb) : (hbuf + (kb - H_DIMC));
                v16bf b = *(const v16bf*)(bsrc + koffB);   // replicated vector B
                c = bf16_mma(a, b, c);
            }
            if (lane == 0 || lane == 16) {
                const int base = (lane == 0) ? 0 : 8;
                for (int r = 0; r < 8; ++r) red[wave][base + r] = c[r];
            }
            __syncthreads();
            if (tid < 16) {
                float s = 0.f;
                for (int w = 0; w < 8; ++w) s += red[w][tid];
                const int row = mbase + tid;
                float u = s + l1b[row];
                u1[row] = (bf16_t)(u > 0.f ? u : 0.f);
            }
            __syncthreads();
        }
        grid_barrier(cnt, gen, nwg);

        // ---- B: l2 + residual + relu ----
        for (int tile = blockIdx.x; tile < H2C / 16; tile += nwg) {
            const int mbase = tile * 16;
            const bf16_t* arow = l2w + (size_t)(mbase + mrow) * H2C;
            v8f c = {};
            for (int kb = kb0; kb < kb0 + H2C / 8; kb += 32) {
                v16bf a = load_a_tile(arow, kb, koffA);
                v16bf b = *(const v16bf*)(u1 + kb + koffB);
                c = bf16_mma(a, b, c);
            }
            if (lane == 0 || lane == 16) {
                const int base = (lane == 0) ? 0 : 8;
                for (int r = 0; r < 8; ++r) red[wave][base + r] = c[r];
            }
            __syncthreads();
            if (tid < 16) {
                float s = 0.f;
                for (int w = 0; w < 8; ++w) s += red[w][tid];
                const int row = mbase + tid;
                const float y  = s + l2b[row];
                const float xv = (row < H_DIMC) ? (float)xivec[row]
                                                : (float)hbuf[row - H_DIMC];
                const float tv = xv + y;
                tmpv[row] = (bf16_t)(tv > 0.f ? tv : 0.f);
            }
            __syncthreads();
        }
        grid_barrier(cnt, gen, nwg);

        // ---- C: xh2h -> h_new ----
        for (int tile = blockIdx.x; tile < H_DIMC / 16; tile += nwg) {
            const int mbase = tile * 16;
            const bf16_t* arow = xhw + (size_t)(mbase + mrow) * H2C;
            v8f c = {};
            for (int kb = kb0; kb < kb0 + H2C / 8; kb += 32) {
                v16bf a = load_a_tile(arow, kb, koffA);
                v16bf b = *(const v16bf*)(tmpv + kb + koffB);
                c = bf16_mma(a, b, c);
            }
            if (lane == 0 || lane == 16) {
                const int base = (lane == 0) ? 0 : 8;
                for (int r = 0; r < 8; ++r) red[wave][base + r] = c[r];
            }
            __syncthreads();
            if (tid < 16) {
                float s = 0.f;
                for (int w = 0; w < 8; ++w) s += red[w][tid];
                const int row = mbase + tid;
                const float hn = s + xhb[row];
                const bf16_t hb = (bf16_t)hn;
                hbuf[row] = hb;
                hall[(size_t)t * H_DIMC + row] = hb;
                if (t == T_STEPS - 1) hfinal[row] = hn;
            }
            __syncthreads();
        }
        grid_barrier(cnt, gen, nwg);
    }
}

// ---------------------------------------------------------------------------
// phase 3: ys[t, o] = out_w[o,:] . h_new_t + out_b[o]   (batched GEMM, WMMA)
// ---------------------------------------------------------------------------
__global__ __launch_bounds__(256) void out_gemm(const bf16_t* __restrict__ wbf,
                                                const bf16_t* __restrict__ hall,
                                                const float*  __restrict__ bias,
                                                float* __restrict__ yout) {
    const int lane  = threadIdx.x & 31;
    const int gw    = (blockIdx.x * blockDim.x + threadIdx.x) >> 5;
    const int nwv   = (gridDim.x * blockDim.x) >> 5;
    const int mrow  = lane & 15;
    const int koffA = (lane & 16) ? 8 : 0;
    const int koffB = (lane & 16) ? 16 : 0;
    constexpr int MT = OUT_DIMC / 16, NT = T_STEPS / 16;

    for (int tile = gw; tile < MT * NT; tile += nwv) {
        const int mt = tile / NT, nt = tile % NT;
        const bf16_t* arow = wbf  + (size_t)(mt * 16 + mrow) * H_DIMC;
        const bf16_t* bcol = hall + (size_t)(nt * 16 + mrow) * H_DIMC;
        v8f c = {};
        for (int kb = 0; kb < H_DIMC; kb += 32) {
            v16bf a = load_a_tile(arow, kb, koffA);
            v16bf b = *(const v16bf*)(bcol + kb + koffB);
            c = bf16_mma(a, b, c);
        }
        const int tcol = nt * 16 + (lane & 15);
        const int o0   = mt * 16 + ((lane & 16) ? 8 : 0);
        float* dst = yout + (size_t)tcol * OUT_DIMC + o0;
        v4f a0, a1;
        for (int r = 0; r < 4; ++r) a0[r] = c[r]     + bias[o0 + r];
        for (int r = 0; r < 4; ++r) a1[r] = c[4 + r] + bias[o0 + 4 + r];
        *(v4f*)(dst)     = a0;
        *(v4f*)(dst + 4) = a1;
    }
}

// ---------------------------------------------------------------------------
extern "C" void kernel_launch(void* const* d_in, const int* in_sizes, int n_in,
                              void* d_out, int out_size, void* d_ws, size_t ws_size,
                              hipStream_t stream) {
    (void)in_sizes; (void)n_in; (void)out_size; (void)ws_size;
    const float* x      = (const float*)d_in[0];
    const float* h      = (const float*)d_in[1];
    const float* inp_w  = (const float*)d_in[2];
    const float* inp_b  = (const float*)d_in[3];
    const float* l1_w   = (const float*)d_in[4];
    const float* l1_b   = (const float*)d_in[5];
    const float* l2_w   = (const float*)d_in[6];
    const float* l2_b   = (const float*)d_in[7];
    const float* xh2h_w = (const float*)d_in[8];
    const float* xh2h_b = (const float*)d_in[9];
    const float* out_w  = (const float*)d_in[10];
    const float* out_b  = (const float*)d_in[11];

    char* ws = (char*)d_ws;
    bf16_t*   inpw_bf = (bf16_t*)(ws + OFF_INPW);
    bf16_t*   l1w_bf  = (bf16_t*)(ws + OFF_L1W);
    bf16_t*   l2w_bf  = (bf16_t*)(ws + OFF_L2W);
    bf16_t*   xhw_bf  = (bf16_t*)(ws + OFF_XHW);
    bf16_t*   outw_bf = (bf16_t*)(ws + OFF_OUTW);
    bf16_t*   x_bf    = (bf16_t*)(ws + OFF_X);
    bf16_t*   xi_bf   = (bf16_t*)(ws + OFF_XI);
    bf16_t*   hall    = (bf16_t*)(ws + OFF_HALL);
    bf16_t*   u1      = (bf16_t*)(ws + OFF_U1);
    bf16_t*   tmpv    = (bf16_t*)(ws + OFF_TMP);
    bf16_t*   hbuf    = (bf16_t*)(ws + OFF_HB);
    unsigned* sync    = (unsigned*)(ws + OFF_SYNC);

    float* ys     = (float*)d_out;                                   // [T, OUT]
    float* hfinal = (float*)d_out + (size_t)T_STEPS * OUT_DIMC;      // [2048]

    // phase 0: convert weights & x to bf16 (L2-resident working set: ~88 MB)
    cvt_kernel<<<1024, 256, 0, stream>>>(inp_w,  inpw_bf, H_DIMC * IN_DIMC);
    cvt_kernel<<<2048, 256, 0, stream>>>(l1_w,   l1w_bf,  H2C * H2C);
    cvt_kernel<<<2048, 256, 0, stream>>>(l2_w,   l2w_bf,  H2C * H2C);
    cvt_kernel<<<2048, 256, 0, stream>>>(xh2h_w, xhw_bf,  H_DIMC * H2C);
    cvt_kernel<<<1024, 256, 0, stream>>>(out_w,  outw_bf, OUT_DIMC * H_DIMC);
    cvt_kernel<<<512,  256, 0, stream>>>(x,      x_bf,    T_STEPS * IN_DIMC);
    init_kernel<<<8, 256, 0, stream>>>(h, hbuf, sync);

    // phase 1: XI = inp_w @ X^T + inp_b  (batched over all timesteps)
    xi_gemm<<<512, 256, 0, stream>>>(inpw_bf, x_bf, inp_b, xi_bf);

    // phase 2: sequential scan (persistent, grid barriers)
    scan_kernel<<<SCAN_WGS, 256, 0, stream>>>(l1w_bf, l2w_bf, xhw_bf,
                                              l1_b, l2_b, xh2h_b,
                                              xi_bf, hall, u1, tmpv, hbuf,
                                              hfinal, sync);

    // phase 3: ys = out_w @ H^T + out_b  (batched over all timesteps)
    out_gemm<<<256, 256, 0, stream>>>(outw_bf, hall, out_b, ys);
}
